// FullyConnected_11055245820569
// MI455X (gfx1250) — matrix operand order repaired and checked by reading, not compiled, not run on hardware
//
#include <hip/hip_runtime.h>

// MAM fully-connected for MI455X (gfx1250), wave32 + WMMA.
// out[m,n] = sum over 2 K-splits of (max_blk + min_blk of 32-wide block partial sums) + bias[n]
// Strategy: convert x,weight to f16 once into workspace (both fit in 192MB L2),
// then one v_wmma_f32_16x16x32_f16 per (16x16 tile, 32-K-block) with C=0 gives the
// exact block partial; running v_max/v_min accumulate the MAM reduction.

typedef _Float16 f16;
typedef __attribute__((ext_vector_type(4)))  _Float16 v4h;
typedef __attribute__((ext_vector_type(8)))  _Float16 v8h;
typedef __attribute__((ext_vector_type(16))) _Float16 v16h;
typedef __attribute__((ext_vector_type(8)))  float    v8f;

static constexpr int M_DIM = 2048;
static constexpr int K_DIM = 4096;
static constexpr int N_DIM = 4096;
static constexpr int ACCBLOCK = 32;                         // reference ACCBLOCK
static constexpr int KSPLIT = K_DIM / 2;                    // reference SPLITS=2 -> 2048 each
static constexpr int BLOCKS_PER_SPLIT = KSPLIT / ACCBLOCK;  // 64

// ---------------- f32 -> f16 conversion (vectorized x4) ----------------
__global__ void cvt_f32_to_f16_x4(const float* __restrict__ s,
                                  f16* __restrict__ d, int n4) {
  int i = blockIdx.x * blockDim.x + threadIdx.x;
  if (i < n4) {
    float4 v = reinterpret_cast<const float4*>(s)[i];
    v4h h = { (f16)v.x, (f16)v.y, (f16)v.z, (f16)v.w };
    reinterpret_cast<v4h*>(d)[i] = h;
  }
}

// ---------------- helpers ----------------
__device__ __forceinline__ v8f splat8(float v) {
  v8f r;
#pragma unroll
  for (int i = 0; i < 8; ++i) r[i] = v;
  return r;
}

__device__ __forceinline__ v16h cat8(v8h lo, v8h hi) {
  return __builtin_shufflevector(lo, hi, 0, 1, 2, 3, 4, 5, 6, 7,
                                         8, 9, 10, 11, 12, 13, 14, 15);
}

// A fragment (16x32 f16): per documented layout, lane(<16) VGPR0..3 = K 0..7,
// VGPR4..7 = K 16..23; lanes>=16 shift K by +8. Caller pre-applies the per-lane
// +8 half offset; the two contiguous 16B chunks sit at +0 and +16 halves.
__device__ __forceinline__ v16h load_a_frag(const f16* p) {
  v8h lo = *reinterpret_cast<const v8h*>(p);
  v8h hi = *reinterpret_cast<const v8h*>(p + 16);
  return cat8(lo, hi);
}

// B fragment (32x16 f16) from [N,K] row-major: same K layout as A; lane(<16)
// holds K 0..7 and 16..23, lanes>=16 shift K by +8 (per-lane +8 half offset
// applied by caller); two 16B chunks at +0 and +16 halves.
__device__ __forceinline__ v16h load_b_frag(const f16* p) {
  v8h lo = *reinterpret_cast<const v8h*>(p);
  v8h hi = *reinterpret_cast<const v8h*>(p + 16);
  return cat8(lo, hi);
}

__device__ __forceinline__ void vmaxmin(v8f& mx, v8f& mn, v8f c) {
#pragma unroll
  for (int i = 0; i < 8; ++i) {
    mx[i] = __builtin_fmaxf(mx[i], c[i]);
    mn[i] = __builtin_fminf(mn[i], c[i]);
  }
}

// ---------------- main MAM-GEMM kernel ----------------
// 256 threads = 8 wave32, arranged 4(M) x 2(N); each wave computes a 32x32
// output macro-tile as 2x2 WMMA 16x16 tiles. Workgroup tile: 128(M) x 64(N).
__launch_bounds__(256)
__global__ void mam_wmma_kernel(const f16* __restrict__ xh,
                                const f16* __restrict__ wh,
                                const float* __restrict__ bias,
                                float* __restrict__ out) {
  const int lane  = threadIdx.x & 31;
  const int wave  = threadIdx.x >> 5;
  const int wm    = wave & 3;   // 0..3 -> M sub-tile
  const int wn    = wave >> 2;  // 0..1 -> N sub-tile
  const int lrow  = lane & 15;
  const int lhalf = lane >> 4;

  const int m0 = blockIdx.y * 128 + wm * 32;
  const int n0 = blockIdx.x * 64  + wn * 32;

  // Per-lane fragment base pointers (row-major [row, K] f16 buffers).
  const f16* pa0 = xh + (size_t)(m0 + lrow) * K_DIM + lhalf * 8;
  const f16* pa1 = pa0 + (size_t)16 * K_DIM;
  const f16* pb0 = wh + (size_t)(n0 + lrow) * K_DIM + lhalf * 8;
  const f16* pb1 = pb0 + (size_t)16 * K_DIM;

  const v8f  zero = splat8(0.0f);
  const float BIG = 3.402823466e+38f;  // finfo(f32).max, as in reference

  v8f acc00 = zero, acc01 = zero, acc10 = zero, acc11 = zero;

#pragma unroll
  for (int split = 0; split < 2; ++split) {
    const f16* a0 = pa0 + (size_t)split * KSPLIT;
    const f16* a1 = pa1 + (size_t)split * KSPLIT;
    const f16* b0 = pb0 + (size_t)split * KSPLIT;
    const f16* b1 = pb1 + (size_t)split * KSPLIT;

    v8f mx00 = splat8(-BIG), mn00 = splat8(BIG);
    v8f mx01 = splat8(-BIG), mn01 = splat8(BIG);
    v8f mx10 = splat8(-BIG), mn10 = splat8(BIG);
    v8f mx11 = splat8(-BIG), mn11 = splat8(BIG);

    for (int blk = 0; blk < BLOCKS_PER_SPLIT; ++blk) {
      // Stream prefetch 16 K-blocks (1 KiB) ahead; operands are L2-resident.
      __builtin_prefetch(a0 + 16 * ACCBLOCK, 0, 0);
      __builtin_prefetch(a1 + 16 * ACCBLOCK, 0, 0);
      __builtin_prefetch(b0 + 16 * ACCBLOCK, 0, 0);
      __builtin_prefetch(b1 + 16 * ACCBLOCK, 0, 0);

      v16h A0 = load_a_frag(a0);
      v16h A1 = load_a_frag(a1);
      v16h B0 = load_b_frag(b0);
      v16h B1 = load_b_frag(b1);

      // One WMMA with C=0 == exact 32-wide block partial sum for a 16x16 tile.
      v8f c00 = __builtin_amdgcn_wmma_f32_16x16x32_f16(false, A0, false, B0,
                                                       (short)0, zero, false, false);
      v8f c01 = __builtin_amdgcn_wmma_f32_16x16x32_f16(false, A0, false, B1,
                                                       (short)0, zero, false, false);
      v8f c10 = __builtin_amdgcn_wmma_f32_16x16x32_f16(false, A1, false, B0,
                                                       (short)0, zero, false, false);
      v8f c11 = __builtin_amdgcn_wmma_f32_16x16x32_f16(false, A1, false, B1,
                                                       (short)0, zero, false, false);

      vmaxmin(mx00, mn00, c00);
      vmaxmin(mx01, mn01, c01);
      vmaxmin(mx10, mn10, c10);
      vmaxmin(mx11, mn11, c11);

      a0 += ACCBLOCK; a1 += ACCBLOCK; b0 += ACCBLOCK; b1 += ACCBLOCK;
    }

    acc00 += mx00 + mn00;
    acc01 += mx01 + mn01;
    acc10 += mx10 + mn10;
    acc11 += mx11 + mn11;
  }

  // Epilogue: C/D layout -> row = r + 8*lhalf (+tile), col = lrow (+tile).
  const float bv0 = bias[n0 + lrow];
  const float bv1 = bias[n0 + 16 + lrow];
  const size_t rbase = (size_t)(m0 + 8 * lhalf) * N_DIM;
#pragma unroll
  for (int r = 0; r < 8; ++r) {
    const size_t row0 = rbase + (size_t)r * N_DIM;
    const size_t row1 = row0 + (size_t)16 * N_DIM;
    out[row0 + n0 + lrow]      = acc00[r] + bv0;
    out[row0 + n0 + 16 + lrow] = acc01[r] + bv1;
    out[row1 + n0 + lrow]      = acc10[r] + bv0;
    out[row1 + n0 + 16 + lrow] = acc11[r] + bv1;
  }
}

// ---------------- host launch ----------------
extern "C" void kernel_launch(void* const* d_in, const int* in_sizes, int n_in,
                              void* d_out, int out_size, void* d_ws, size_t ws_size,
                              hipStream_t stream) {
  (void)in_sizes; (void)n_in; (void)out_size; (void)ws_size;

  const float* x      = (const float*)d_in[0];  // [M, K] f32
  const float* weight = (const float*)d_in[1];  // [N, K] f32
  const float* bias   = (const float*)d_in[2];  // [N]    f32
  float*       out    = (float*)d_out;          // [M, N] f32

  // Workspace layout: xh f16 [M,K] (16 MB) | wh f16 [N,K] (32 MB)
  f16* xh = (f16*)d_ws;
  f16* wh = xh + (size_t)M_DIM * K_DIM;

  {
    const int nx4 = (M_DIM * K_DIM) / 4;
    cvt_f32_to_f16_x4<<<(nx4 + 255) / 256, 256, 0, stream>>>(x, xh, nx4);
    const int nw4 = (N_DIM * K_DIM) / 4;
    cvt_f32_to_f16_x4<<<(nw4 + 255) / 256, 256, 0, stream>>>(weight, wh, nw4);
  }

  dim3 grid(N_DIM / 64, M_DIM / 128);  // (64, 16)
  mam_wmma_kernel<<<grid, 256, 0, stream>>>(xh, wh, bias, out);
}
